// BottleneckModel_33947421507775
// MI455X (gfx1250) — matrix operand order repaired and checked by reading, not compiled, not run on hardware
//
#include <hip/hip_runtime.h>
#include <math.h>

typedef __bf16 bf16_t;
typedef __attribute__((ext_vector_type(16))) __bf16 v16bf;
typedef __attribute__((ext_vector_type(8)))  float  v8f;
typedef __attribute__((ext_vector_type(4))) unsigned int v4u;
typedef __attribute__((ext_vector_type(8))) int v8i;
typedef __attribute__((ext_vector_type(4))) int v4i;

#define B_    16
#define L_    64
#define P_    512
#define H_    16
#define DH_   32
#define DIM_  128
#define CDIM_ 256
#define INNER_ 512

#if defined(__has_builtin)
#  if __has_builtin(__builtin_amdgcn_tensor_load_to_lds) && __has_builtin(__builtin_amdgcn_s_wait_tensorcnt)
#    define USE_TDM 1
#  else
#    define USE_TDM 0
#  endif
#else
#  define USE_TDM 0
#endif

// A-fragment k index for 16x32 bf16 A (row m = lane&15)
__device__ __forceinline__ int kA(int e, int g) {
  return ((e >> 3) << 4) + (g << 3) + (e & 7);
}
// B-fragment k index for 32x16 bf16 B (col n = lane&15)
__device__ __forceinline__ int kB(int e, int g) {
  return ((e >> 3) << 4) + (g << 3) + (e & 7);
}

__device__ __forceinline__ v8f zero8() {
  v8f z;
  #pragma unroll
  for (int i = 0; i < 8; ++i) z[i] = 0.0f;
  return z;
}

__device__ __forceinline__ v8f wmma_bf16(v16bf a, v16bf b, v8f c) {
  return __builtin_amdgcn_wmma_f32_16x16x32_bf16(false, a, false, b, (short)0, c, false, false);
}

__device__ __forceinline__ float mishf(float x) {
  float sp = (x > 20.0f) ? x : log1pf(__expf(x));
  return x * tanhf(sp);
}

#if USE_TDM
// TDM: load a 3D tile [z=16][y=16][x=32] of 2-byte elements into contiguous LDS.
// x contiguous (32 elems), y stride = 32 elems, z stride = zstride elems.
__device__ __forceinline__ void tdm_load_tile3d(unsigned int lds_off, const void* gaddr,
                                                unsigned int ydim, unsigned int zstride) {
  unsigned long long ga = (unsigned long long)(uintptr_t)gaddr;
  v4u g0;
  g0[0] = 1u;                                    // count=1 (valid user descriptor)
  g0[1] = lds_off;                               // lds_addr (bytes)
  g0[2] = (unsigned int)ga;                      // global_addr[31:0]
  g0[3] = (unsigned int)((ga >> 32) & 0x01ffffffu) | (2u << 30); // addr[56:32] | type=2
  v8i g1;
  g1[0] = 0x00010000;                            // workgroup_mask=0, data_size=1 (2 bytes)
  g1[1] = (int)(32u << 16);                      // tensor_dim0 = 32 (bits 79:48 low part)
  g1[2] = (int)(ydim << 16);                     // tensor_dim0 hi=0 | tensor_dim1[15:0]
  g1[3] = (int)((ydim >> 16) | (32u << 16));     // tensor_dim1 hi | tile_dim0 = 32
  g1[4] = (int)(16u | (16u << 16));              // tile_dim1 = 16, tile_dim2 = 16
  g1[5] = 32;                                    // tensor_dim0_stride[31:0] = 32
  g1[6] = (int)(zstride << 16);                  // dim0_stride hi=0 | dim1_stride[15:0]
  g1[7] = (int)(zstride >> 16);                  // dim1_stride[47:16]
  v4i g2;
  g2[0] = 16;                                    // tensor_dim2 = 16 (z extent)
  g2[1] = 0; g2[2] = 0; g2[3] = 0;               // tensor_dim3=0, dim2_stride=0, tile_dim3=0
  v4i g3;
  g3[0] = 0; g3[1] = 0; g3[2] = 0; g3[3] = 0;
#if __clang_major__ >= 23
  v8i g4;
  #pragma unroll
  for (int i = 0; i < 8; ++i) g4[i] = 0;
  __builtin_amdgcn_tensor_load_to_lds(g0, g1, g2, g3, g4, 0);
#else
  __builtin_amdgcn_tensor_load_to_lds(g0, g1, g2, g3, 0);
#endif
}
#endif

// ---------------- Kernel 1: qk = hl @ w_qk, scaled, bf16 [B,H,L,DH] ----------------
__global__ void proj_qk_kernel(const float* __restrict__ hl,
                               const float* __restrict__ w,
                               bf16_t* __restrict__ qk) {
  int lane = threadIdx.x & 31;
  int l = lane & 15, g = lane >> 4;
  int wid = blockIdx.x * (blockDim.x >> 5) + (threadIdx.x >> 5); // 2048 waves
  int tile_n = wid & 31;   // 32 n-tiles of INNER
  int tile_m = wid >> 5;   // 64 m-tiles of B*L
  v8f acc = zero8();
  for (int kk = 0; kk < DIM_ / 32; ++kk) {
    v16bf a, bm;
    #pragma unroll
    for (int e = 0; e < 16; ++e) {
      a[e]  = (bf16_t)hl[(tile_m * 16 + l) * DIM_ + kk * 32 + kA(e, g)];
      bm[e] = (bf16_t)w[(kk * 32 + kB(e, g)) * INNER_ + tile_n * 16 + l];
    }
    acc = wmma_bf16(a, bm, acc);
  }
  const float scale = 0.17677669529663687f; // DH^-0.5 folded into qk
  #pragma unroll
  for (int r = 0; r < 8; ++r) {
    int row = tile_m * 16 + r + 8 * g;
    int bb = row >> 6, lr = row & 63;
    int col = tile_n * 16 + l;
    int h = col >> 5, d = col & 31;
    qk[(((bb * H_ + h) * L_ + lr) << 5) + d] = (bf16_t)(acc[r] * scale);
  }
}

// ------- Kernel 2: ctx @ W -> bf16; mode 0: [B,H,P,DH] (cqk); mode 1: [B,H,DH,P] (cv^T) -------
__global__ void proj_ctx_kernel(const float* __restrict__ ctx,
                                const float* __restrict__ w,
                                bf16_t* __restrict__ out, int mode) {
  int lane = threadIdx.x & 31;
  int l = lane & 15, g = lane >> 4;
  int wid = blockIdx.x * (blockDim.x >> 5) + (threadIdx.x >> 5); // 16384 waves
  int tile_n = wid & 31;   // 32 n-tiles
  int tile_m = wid >> 5;   // 512 m-tiles of B*P
  v8f acc = zero8();
  for (int kk = 0; kk < CDIM_ / 32; ++kk) {
    if (kk + 1 < CDIM_ / 32)
      __builtin_prefetch(&ctx[(tile_m * 16 + l) * CDIM_ + (kk + 1) * 32], 0, 0);
    v16bf a, bm;
    #pragma unroll
    for (int e = 0; e < 16; ++e) {
      a[e]  = (bf16_t)ctx[(tile_m * 16 + l) * CDIM_ + kk * 32 + kA(e, g)];
      bm[e] = (bf16_t)w[(kk * 32 + kB(e, g)) * INNER_ + tile_n * 16 + l];
    }
    acc = wmma_bf16(a, bm, acc);
  }
  #pragma unroll
  for (int r = 0; r < 8; ++r) {
    int row = tile_m * 16 + r + 8 * g;
    int bb = row >> 9, p = row & 511;
    int col = tile_n * 16 + l;
    int h = col >> 5, d = col & 31;
    if (mode == 0)
      out[(((bb * H_ + h) * P_ + p) << 5) + d] = (bf16_t)acc[r];
    else
      out[(((bb * H_ + h) * DH_ + d) << 9) + p] = (bf16_t)acc[r];
  }
}

// ---- Kernel 3: scores[b][2H][L][P] = MLP(concat(qk.cqk^T, PE-MLP(dis))), pre-softmax ----
__global__ void scores_kernel(const bf16_t* __restrict__ qk,
                              const bf16_t* __restrict__ cqk,
                              const float* __restrict__ xp,
                              const float* __restrict__ xl,
                              const float* __restrict__ w_pe1, const float* __restrict__ b_pe1,
                              const float* __restrict__ w_pe2, const float* __restrict__ b_pe2,
                              const float* __restrict__ sigma,
                              const float* __restrict__ w_mlp1,
                              const float* __restrict__ w_mlp2, const float* __restrict__ b_mlp2,
                              float* __restrict__ scores) {
  __shared__ bf16_t sc[256][32];       // pair (l*16+p) x channel
  __shared__ bf16_t hid[4][16][64];    // per-wave layer1 activations
  __shared__ float pw1[192], pb1[64], pw2[1024], pb2[16];
  __shared__ float xpS[16][3], xlS[16][3];
#if USE_TDM
  __shared__ bf16_t qkS[16 * 16 * 32]; // [h][l][d] staged by TDM
  __shared__ bf16_t cqS[16 * 16 * 32]; // [h][p][d] staged by TDM
#endif

  int b = blockIdx.z, lt = blockIdx.y, pt = blockIdx.x;
  int l0 = lt * 16, p0 = pt * 16;
  int tid = threadIdx.x;
  int wv = tid >> 5, lane = tid & 31, l = lane & 15, g = lane >> 4;

#if USE_TDM
  // stage 0a: wave 0 kicks the Tensor Data Mover for both input tiles
  if (wv == 0) {
    tdm_load_tile3d((unsigned int)(uintptr_t)qkS,
                    qk + (((size_t)b * H_ * L_ + l0) << 5),
                    (unsigned int)L_, (unsigned int)(L_ * DH_));
    tdm_load_tile3d((unsigned int)(uintptr_t)cqS,
                    cqk + (((size_t)b * H_ * P_ + p0) << 5),
                    (unsigned int)P_, (unsigned int)(P_ * DH_));
  }
#endif

  // stage 0b: PE weights + coord tiles into LDS
  for (int i = tid; i < 192; i += 128) pw1[i] = w_pe1[i];
  for (int i = tid; i < 64; i += 128) pb1[i] = b_pe1[i];
  for (int i = tid; i < 1024; i += 128) pw2[i] = w_pe2[i];
  if (tid < 16) pb2[tid] = b_pe2[tid];
  if (tid < 48) xpS[tid / 3][tid % 3] = xp[(b * P_ + p0 + tid / 3) * 3 + tid % 3];
  if (tid >= 64 && tid < 112) {
    int t = tid - 64;
    xlS[t / 3][t % 3] = xl[(b * L_ + l0 + t / 3) * 3 + t % 3];
  }
#if USE_TDM
  if (wv == 0) __builtin_amdgcn_s_wait_tensorcnt(0);
#endif
  __syncthreads();

  // stage 1a: sim heads (scale already folded into qk); one WMMA per head
  for (int i = 0; i < 4; ++i) {
    int h = wv * 4 + i;
    v16bf a, bm;
    #pragma unroll
    for (int e = 0; e < 16; ++e) {
#if USE_TDM
      a[e]  = qkS[((h * 16 + l) << 5) + kA(e, g)];
      bm[e] = cqS[((h * 16 + l) << 5) + kB(e, g)];
#else
      a[e]  = qk[(((b * H_ + h) * L_ + l0 + l) << 5) + kA(e, g)];
      bm[e] = cqk[(((b * H_ + h) * P_ + p0 + l) << 5) + kB(e, g)];
#endif
    }
    v8f acc = wmma_bf16(a, bm, zero8());
    #pragma unroll
    for (int r = 0; r < 8; ++r)
      sc[(r + 8 * g) * 16 + l][h] = (bf16_t)acc[r];
  }

  // stage 1b: positional-encoding MLP (3 -> 64 -> 16), 2 pairs per thread
  float sg = sigma[0];
  float cexp = -0.5f / (sg * sg);
  for (int j = 0; j < 2; ++j) {
    int q = tid * 2 + j;
    int lr = q >> 4, pc = q & 15;
    float pe[3];
    #pragma unroll
    for (int c = 0; c < 3; ++c)
      pe[c] = __expf((xpS[pc][c] - xlS[lr][c]) * cexp);
    float emb[16];
    #pragma unroll
    for (int hh = 0; hh < 16; ++hh) emb[hh] = pb2[hh];
    for (int jj = 0; jj < 64; ++jj) {
      float hv = pb1[jj];
      #pragma unroll
      for (int c = 0; c < 3; ++c) hv += pe[c] * pw1[c * 64 + jj];
      hv = mishf(hv);
      #pragma unroll
      for (int hh = 0; hh < 16; ++hh) emb[hh] += hv * pw2[jj * 16 + hh];
    }
    #pragma unroll
    for (int hh = 0; hh < 16; ++hh) sc[q][16 + hh] = (bf16_t)emb[hh];
  }
  __syncthreads();

  // stage 2: attn MLP 2H -> 4H (mish) -> 2H via WMMA over 16-pair tiles
  for (int i = 0; i < 4; ++i) {
    int q0 = (wv * 4 + i) * 16;
    v16bf a;
    #pragma unroll
    for (int e = 0; e < 16; ++e) a[e] = sc[q0 + l][kA(e, g)];
    for (int nt = 0; nt < 4; ++nt) {
      v16bf bm;
      #pragma unroll
      for (int e = 0; e < 16; ++e)
        bm[e] = (bf16_t)w_mlp1[kB(e, g) * 64 + nt * 16 + l];
      v8f acc = wmma_bf16(a, bm, zero8());
      #pragma unroll
      for (int r = 0; r < 8; ++r)
        hid[wv][r + 8 * g][nt * 16 + l] = (bf16_t)mishf(acc[r]);
    }
    for (int nt2 = 0; nt2 < 2; ++nt2) {
      v8f acc = zero8();
      for (int kk = 0; kk < 2; ++kk) {
        v16bf a2, bm;
        #pragma unroll
        for (int e = 0; e < 16; ++e) {
          a2[e] = hid[wv][l][kk * 32 + kA(e, g)];
          bm[e] = (bf16_t)w_mlp2[(kk * 32 + kB(e, g)) * 32 + nt2 * 16 + l];
        }
        acc = wmma_bf16(a2, bm, acc);
      }
      int ch = nt2 * 16 + l;
      float bias = b_mlp2[ch];
      #pragma unroll
      for (int r = 0; r < 8; ++r) {
        int q = q0 + r + 8 * g;
        int lr = q >> 4, pc = q & 15;
        scores[((b * 32 + ch) * 64 + l0 + lr) * 512 + p0 + pc] = acc[r] + bias;
      }
    }
  }
}

// ---------------- Kernel 4: in-place softmax over P per (b, ch, l) row ----------------
__global__ void softmax_kernel(float* __restrict__ s) {
  int wv = threadIdx.x >> 5, lane = threadIdx.x & 31;
  int row = blockIdx.x * 8 + wv; // 32768 rows
  float* p = s + (size_t)row * 512;
  float v[16];
  float mx = -3.4e38f;
  #pragma unroll
  for (int i = 0; i < 16; ++i) { v[i] = p[i * 32 + lane]; mx = fmaxf(mx, v[i]); }
  #pragma unroll
  for (int off = 16; off >= 1; off >>= 1) mx = fmaxf(mx, __shfl_xor(mx, off));
  float sum = 0.0f;
  #pragma unroll
  for (int i = 0; i < 16; ++i) { v[i] = __expf(v[i] - mx); sum += v[i]; }
  #pragma unroll
  for (int off = 16; off >= 1; off >>= 1) sum += __shfl_xor(sum, off);
  float inv = 1.0f / sum;
  #pragma unroll
  for (int i = 0; i < 16; ++i) p[i * 32 + lane] = v[i] * inv;
}

// ---------- Kernel 5: out = (attn[:, :H] @ cv) @ w_out + b_out, fused through LDS ----------
__global__ void outproj_kernel(const float* __restrict__ scores,
                               const bf16_t* __restrict__ cvT,
                               const float* __restrict__ w_out,
                               const float* __restrict__ b_out,
                               float* __restrict__ out) {
  __shared__ bf16_t ov[16][512]; // attn@cv intermediate (16 ligand rows x inner)
  int b = blockIdx.y, l0 = blockIdx.x * 16;
  int tid = threadIdx.x;
  int wv = tid >> 5, lane = tid & 31, l = lane & 15, g = lane >> 4;

  for (int i = 0; i < 4; ++i) {
    int h = wv * 4 + i;
    v8f acc0 = zero8(), acc1 = zero8();
    for (int kk = 0; kk < 16; ++kk) {
      v16bf a, b0, b1;
      #pragma unroll
      for (int e = 0; e < 16; ++e) {
        a[e]  = (bf16_t)scores[((b * 32 + h) * 64 + l0 + l) * 512 + kk * 32 + kA(e, g)];
        b0[e] = cvT[(((b * H_ + h) * DH_ + l) << 9) + kk * 32 + kB(e, g)];
        b1[e] = cvT[(((b * H_ + h) * DH_ + 16 + l) << 9) + kk * 32 + kB(e, g)];
      }
      acc0 = wmma_bf16(a, b0, acc0);
      acc1 = wmma_bf16(a, b1, acc1);
    }
    #pragma unroll
    for (int r = 0; r < 8; ++r) {
      int m = r + 8 * g;
      ov[m][h * 32 + l]      = (bf16_t)acc0[r];
      ov[m][h * 32 + 16 + l] = (bf16_t)acc1[r];
    }
  }
  __syncthreads();

  for (int i = 0; i < 2; ++i) {
    int nt = wv * 2 + i;
    v8f acc = zero8();
    for (int kk = 0; kk < 16; ++kk) {
      v16bf a, bm;
      #pragma unroll
      for (int e = 0; e < 16; ++e) {
        a[e]  = ov[l][kk * 32 + kA(e, g)];
        bm[e] = (bf16_t)w_out[(kk * 32 + kB(e, g)) * DIM_ + nt * 16 + l];
      }
      acc = wmma_bf16(a, bm, acc);
    }
    float bias = b_out[nt * 16 + l];
    #pragma unroll
    for (int r = 0; r < 8; ++r)
      out[(b * 64 + l0 + r + 8 * g) * DIM_ + nt * 16 + l] = acc[r] + bias;
  }
}

// ------- Kernel 6: distance path; softmax rows sum to 1 so dis-agg = attn@xp - xl -------
__global__ void dispath_kernel(const float* __restrict__ scores,
                               const float* __restrict__ xp,
                               const float* __restrict__ xl,
                               const float* __restrict__ w_dis1, const float* __restrict__ b_dis1,
                               const float* __restrict__ w_dis2, const float* __restrict__ b_dis2,
                               float* __restrict__ out_dis) {
  __shared__ float od[3][16];
  __shared__ float hb[3][32];
  int bl = blockIdx.x;
  int b = bl >> 6, lr = bl & 63;
  int tid = threadIdx.x;
  if (tid < 48) {
    int crd = tid >> 4, h = tid & 15;
    const float* srow = scores + ((size_t)(b * 32 + 16 + h) * 64 + lr) * 512;
    float s = 0.0f;
    for (int p = 0; p < 512; ++p)
      s += srow[p] * xp[(b * P_ + p) * 3 + crd];
    od[crd][h] = s - xl[bl * 3 + crd];
  }
  __syncthreads();
  if (tid < 96) {
    int crd = tid >> 5, j = tid & 31;
    float hv = b_dis1[j];
    #pragma unroll
    for (int i = 0; i < 16; ++i) hv += od[crd][i] * w_dis1[i * 32 + j];
    hb[crd][j] = mishf(hv);
  }
  __syncthreads();
  if (tid < 48) {
    int crd = tid >> 4, h = tid & 15;
    float o = b_dis2[h];
    #pragma unroll
    for (int j = 0; j < 32; ++j) o += hb[crd][j] * w_dis2[j * 16 + h];
    out_dis[(bl * 3 + crd) * 16 + h] = o;
  }
}

extern "C" void kernel_launch(void* const* d_in, const int* in_sizes, int n_in,
                              void* d_out, int out_size, void* d_ws, size_t ws_size,
                              hipStream_t stream) {
  (void)in_sizes; (void)n_in; (void)out_size; (void)ws_size;
  const float* hl     = (const float*)d_in[0];
  const float* ctx    = (const float*)d_in[1];
  const float* xp     = (const float*)d_in[2];
  const float* xl     = (const float*)d_in[3];
  const float* w_qk   = (const float*)d_in[4];
  /* d_in[5] = w_v is unused by the reference */
  const float* w_cqk  = (const float*)d_in[6];
  const float* w_cv   = (const float*)d_in[7];
  const float* w_mlp1 = (const float*)d_in[8];
  const float* w_mlp2 = (const float*)d_in[9];
  const float* b_mlp2 = (const float*)d_in[10];
  const float* w_out  = (const float*)d_in[11];
  const float* b_out  = (const float*)d_in[12];
  const float* w_dis1 = (const float*)d_in[13];
  const float* b_dis1 = (const float*)d_in[14];
  const float* w_dis2 = (const float*)d_in[15];
  const float* b_dis2 = (const float*)d_in[16];
  const float* sigma  = (const float*)d_in[17];
  const float* w_pe1  = (const float*)d_in[18];
  const float* b_pe1  = (const float*)d_in[19];
  const float* w_pe2  = (const float*)d_in[20];
  const float* b_pe2  = (const float*)d_in[21];

  char* ws = (char*)d_ws;
  bf16_t* qk_ws  = (bf16_t*)(ws);                          // 1 MB
  bf16_t* cqk_ws = (bf16_t*)(ws + (size_t)1 * (1u << 20)); // 8 MB
  bf16_t* cvT_ws = (bf16_t*)(ws + (size_t)9 * (1u << 20)); // 8 MB
  float*  scores = (float*)(ws + (size_t)17 * (1u << 20)); // 64 MB
  float* out     = (float*)d_out;
  float* out_dis = out + B_ * L_ * DIM_;

  proj_qk_kernel<<<512, 128, 0, stream>>>(hl, w_qk, qk_ws);
  proj_ctx_kernel<<<2048, 256, 0, stream>>>(ctx, w_cqk, cqk_ws, 0);
  proj_ctx_kernel<<<2048, 256, 0, stream>>>(ctx, w_cv, cvT_ws, 1);
  scores_kernel<<<dim3(32, 4, 16), 128, 0, stream>>>(
      qk_ws, cqk_ws, xp, xl, w_pe1, b_pe1, w_pe2, b_pe2, sigma,
      w_mlp1, w_mlp2, b_mlp2, scores);
  softmax_kernel<<<4096, 256, 0, stream>>>(scores);
  outproj_kernel<<<dim3(4, 16), 128, 0, stream>>>(scores, cvT_ws, w_out, b_out, out);
  dispath_kernel<<<1024, 128, 0, stream>>>(scores, xp, xl,
      w_dis1, b_dis1, w_dis2, b_dis2, out_dis);
}
